// AttentionBlock_23192823398491
// MI455X (gfx1250) — compile-verified, matrix operand order checked by reading, not run on hardware
//
#include <hip/hip_runtime.h>

// ---------- types ----------
typedef __attribute__((ext_vector_type(16))) __bf16 v16bf;
typedef __attribute__((ext_vector_type(8)))  float  v8f;
typedef __attribute__((ext_vector_type(4)))  unsigned int u32x4;
typedef __attribute__((ext_vector_type(8)))  int    i32x8;
typedef __attribute__((ext_vector_type(4)))  int    i32x4;
typedef __attribute__((ext_vector_type(4)))  float  f32x4;

union FragBf {
    v16bf v;
    u32x4 q[2];
};

// round-to-nearest-even float -> bf16 bits
static __device__ __forceinline__ unsigned short f2bf(float f) {
    unsigned int u = __float_as_uint(f);
    u += 0x7FFFu + ((u >> 16) & 1u);
    return (unsigned short)(u >> 16);
}

// ---------- TDM: async 2-D tile load (rows x 32 bf16 cols) into LDS ----------
// D# per CDNA5 ISA ch.8: group0 = {flags, lds_addr, global_addr, type=2},
// group1 = {mask/data_size, dims/strides}, groups 2/3 unused (2-D tile).
// This toolchain exposes the 6-arg builtin:
//   (u32x4 g0, i32x8 g1, i32x4 g2, i32x4 g3, i32x8 g4, i32 cpol)
static __device__ __forceinline__ void tdm_load_tile_b(
        const unsigned short* gsrc,   // element (row0, k0) of Bt
        unsigned int ldsAddr,         // byte offset in LDS
        int rows, int ldbElems) {
    unsigned long long ga = (unsigned long long)(uintptr_t)gsrc;
    u32x4 g0;
    g0[0] = 1u;                                        // count=1, no gather
    g0[1] = ldsAddr;                                   // LDS byte address
    g0[2] = (unsigned int)ga;                          // global_addr[31:0]
    g0[3] = (unsigned int)((ga >> 32) & 0x01FFFFFFu)   // global_addr[56:32]
            | 0x80000000u;                             // type=2 (bits 127:126)

    const unsigned int td0 = 32u;                      // tensor_dim0 (elements)
    const unsigned int td1 = (unsigned int)rows;       // tensor_dim1
    const unsigned long long s0 = (unsigned long long)ldbElems; // dim0 stride

    i32x8 g1;
    g1[0] = (int)(1u << 16);                           // wg_mask=0, data_size=1 (2B)
    g1[1] = (int)((td0 & 0xFFFFu) << 16);              // tensor_dim0[15:0] @ bits 63:48
    g1[2] = (int)((td0 >> 16) | ((td1 & 0xFFFFu) << 16));   // dim0 hi | dim1 lo
    g1[3] = (int)((td1 >> 16) | (32u << 16));          // dim1 hi | tile_dim0=32
    g1[4] = (int)((unsigned int)rows);                 // tile_dim1=rows, tile_dim2=0
    g1[5] = (int)(unsigned int)(s0 & 0xFFFFFFFFull);   // dim0_stride[31:0]
    g1[6] = (int)(unsigned int)((s0 >> 32) & 0xFFFFu); // dim0_stride[47:32]
    g1[7] = 0;
    i32x4 gz4 = {0, 0, 0, 0};
    i32x8 gz8 = {0, 0, 0, 0, 0, 0, 0, 0};
    __builtin_amdgcn_tensor_load_to_lds(g0, g1, gz4, gz4, gz8, 0);
}

// ---------- weight transpose + fp32 -> bf16 ----------
__global__ void transpose_w_kernel(const float* __restrict__ W,
                                   unsigned short* __restrict__ WT) {
    int idx = blockIdx.x * 256 + threadIdx.x;      // 0 .. 512*512-1
    int i = idx >> 9;                              // input channel
    int o = idx & 511;                             // output channel
    WT[(size_t)o * 512 + i] = f2bf(W[(size_t)i * 512 + o]);
}

// ---------- GroupNorm ----------
// one block per (batch, group): 512 blocks, 256 threads; 16384 values each
__global__ void groupnorm_kernel(const float* __restrict__ x,
                                 const float* __restrict__ gamma,
                                 const float* __restrict__ beta,
                                 float* __restrict__ xn_f32,           // residual base
                                 unsigned short* __restrict__ xn_bf) { // bf16 for GEMMs
    const int bg = blockIdx.x;
    const int b  = bg >> 5;
    const int g  = bg & 31;
    const int t  = threadIdx.x;

    float vals[4][16];
    float sum = 0.f, sq = 0.f;
    #pragma unroll
    for (int i = 0; i < 4; ++i) {
        const int s = t + i * 256;
        const float* p = x + ((size_t)(b * 1024 + s)) * 512 + g * 16;
        #pragma unroll
        for (int j = 0; j < 16; j += 4) {
            f32x4 f = *(const f32x4*)(p + j);
            #pragma unroll
            for (int u = 0; u < 4; ++u) {
                float v = f[u];
                vals[i][j + u] = v;
                sum += v;
                sq  += v * v;
            }
        }
    }

    __shared__ float s1[256], s2[256];
    s1[t] = sum; s2[t] = sq;
    __syncthreads();
    for (int off = 128; off > 0; off >>= 1) {
        if (t < off) { s1[t] += s1[t + off]; s2[t] += s2[t + off]; }
        __syncthreads();
    }
    const float mean = s1[0] * (1.0f / 16384.0f);
    const float var  = s2[0] * (1.0f / 16384.0f) - mean * mean;
    const float inv  = rsqrtf(var + 1e-3f);

    float gm[16], bt[16];
    #pragma unroll
    for (int j = 0; j < 16; ++j) {
        gm[j] = gamma[g * 16 + j] * inv;
        bt[j] = beta[g * 16 + j];
    }

    #pragma unroll
    for (int i = 0; i < 4; ++i) {
        const int s = t + i * 256;
        const size_t base = ((size_t)(b * 1024 + s)) * 512 + g * 16;
        #pragma unroll
        for (int j = 0; j < 16; ++j) {
            float y = (vals[i][j] - mean) * gm[j] + bt[j];
            xn_f32[base + j] = y;
            xn_bf[base + j]  = f2bf(y);
        }
    }
}

// ---------- register-blocked bf16 WMMA GEMM with TDM-staged B tile ----------
// C[M][N] = A[M][K] * Bt[N][K]^T   (Bt row-major [N][K])
// OUT_MODE: 0 = bf16 row-major, 1 = f32 * scale, 2 = bf16 transposed, 3 = f32 +=
// grid: (M/128, N/128, batch); 256 threads = 8 waves (2 M x 4 N);
// wave tile 64x32 = 4x2 accumulators. B block tile (128x32) is DMA'd by the
// Tensor Data Mover into double-buffered LDS; A comes straight from global.
template <int OUT_MODE, bool HAS_BIAS>
__global__ void gemm_bf16_wmma(const unsigned short* __restrict__ A, size_t sA, int lda,
                               const unsigned short* __restrict__ Bt, size_t sB, int ldb,
                               void* __restrict__ Cptr, size_t sC, int ldc,
                               const float* __restrict__ bias,
                               float scale, int K) {
    __shared__ __align__(16) unsigned short ldsB[2][128 * 32];   // 16 KB

    const int lane = threadIdx.x & 31;
    const int wave = threadIdx.x >> 5;
    const int wm = wave >> 2;                       // 0..1
    const int wn = wave & 3;                        // 0..3
    const int tileM = blockIdx.x * 128 + wm * 64;
    const size_t z = blockIdx.z;

    const unsigned short* a0 = A + z * sA;
    const unsigned short* btile = Bt + z * sB + (size_t)(blockIdx.y * 128) * ldb;

    const int row  = lane & 15;
    const int half = lane >> 4;

    const unsigned short* ar[4];
    #pragma unroll
    for (int mi = 0; mi < 4; ++mi)
        ar[mi] = a0 + (size_t)(tileM + mi * 16 + row) * lda + half * 8;

    // B fragment source inside the LDS tile: row-local = wn*32 + ni*16 + row
    const unsigned short* bl[2];
    #pragma unroll
    for (int ni = 0; ni < 2; ++ni)
        bl[ni] = &ldsB[0][(wn * 32 + ni * 16 + row) * 32 + half * 8];
    const unsigned int bufStride = 128 * 32;        // elements between buffers

    const unsigned int ldsBase = (unsigned int)(uintptr_t)&ldsB[0][0];

    // prologue: TDM fill of buffer 0
    if (wave == 0) {
        tdm_load_tile_b(btile, ldsBase, 128, ldb);
        __builtin_amdgcn_s_wait_tensorcnt(0);
    }
    __syncthreads();

    v8f acc[4][2] = {};

    for (int k0 = 0; k0 < K; k0 += 32) {
        const int cur  = (k0 >> 5) & 1;
        const bool more = (k0 + 32) < K;
        if (more && wave == 0)                       // async fill of next buffer
            tdm_load_tile_b(btile + (k0 + 32), ldsBase + (cur ^ 1) * bufStride * 2,
                            128, ldb);

        FragBf fa[4], fb[2];
        #pragma unroll
        for (int mi = 0; mi < 4; ++mi) {
            fa[mi].q[0] = *(const u32x4*)(ar[mi] + k0);
            fa[mi].q[1] = *(const u32x4*)(ar[mi] + k0 + 16);
            __builtin_prefetch(ar[mi] + k0 + 128, 0, 1);   // speculative, OOB-safe
        }
        #pragma unroll
        for (int ni = 0; ni < 2; ++ni) {
            fb[ni].q[0] = *(const u32x4*)(bl[ni] + cur * bufStride);
            fb[ni].q[1] = *(const u32x4*)(bl[ni] + cur * bufStride + 16);
        }
        #pragma unroll
        for (int mi = 0; mi < 4; ++mi)
            #pragma unroll
            for (int ni = 0; ni < 2; ++ni)
                acc[mi][ni] = __builtin_amdgcn_wmma_f32_16x16x32_bf16(
                    false, fa[mi].v, false, fb[ni].v, (short)0, acc[mi][ni],
                    false, false);

        if (more) {
            if (wave == 0) __builtin_amdgcn_s_wait_tensorcnt(0);
            __syncthreads();
        }
    }

    #pragma unroll
    for (int ni = 0; ni < 2; ++ni) {
        const int n = blockIdx.y * 128 + wn * 32 + ni * 16 + (lane & 15);
        const float bval = HAS_BIAS ? bias[n] : 0.0f;
        #pragma unroll
        for (int mi = 0; mi < 4; ++mi) {
            #pragma unroll
            for (int r = 0; r < 8; ++r) {
                const int m = tileM + mi * 16 + r + 8 * half;
                float val = acc[mi][ni][r] + bval;
                if constexpr (OUT_MODE == 0) {
                    ((unsigned short*)Cptr)[z * sC + (size_t)m * ldc + n] = f2bf(val);
                } else if constexpr (OUT_MODE == 1) {
                    ((float*)Cptr)[z * sC + (size_t)m * ldc + n] = val * scale;
                } else if constexpr (OUT_MODE == 2) {
                    ((unsigned short*)Cptr)[z * sC + (size_t)n * ldc + m] = f2bf(val);
                } else {
                    ((float*)Cptr)[z * sC + (size_t)m * ldc + n] += val;
                }
            }
        }
    }
}

// ---------- row softmax: fp32 scores -> bf16 attn ----------
__global__ void softmax_kernel(const float* __restrict__ scores,
                               unsigned short* __restrict__ attn) {
    const size_t rowi = blockIdx.x;
    const float* s = scores + rowi * 1024;
    const int t = threadIdx.x;

    f32x4 v = *(const f32x4*)(s + t * 4);
    float mx = fmaxf(fmaxf(v[0], v[1]), fmaxf(v[2], v[3]));

    __shared__ float red[256];
    red[t] = mx;
    __syncthreads();
    for (int off = 128; off > 0; off >>= 1) {
        if (t < off) red[t] = fmaxf(red[t], red[t + off]);
        __syncthreads();
    }
    mx = red[0];
    __syncthreads();

    float e[4];
    float lsum = 0.f;
    #pragma unroll
    for (int u = 0; u < 4; ++u) {
        e[u] = __expf(v[u] - mx);
        lsum += e[u];
    }
    red[t] = lsum;
    __syncthreads();
    for (int off = 128; off > 0; off >>= 1) {
        if (t < off) red[t] += red[t + off];
        __syncthreads();
    }
    const float inv = 1.0f / red[0];

    unsigned short* o = attn + rowi * 1024 + t * 4;
    #pragma unroll
    for (int u = 0; u < 4; ++u) o[u] = f2bf(e[u] * inv);
}

// ---------- host launch ----------
extern "C" void kernel_launch(void* const* d_in, const int* in_sizes, int n_in,
                              void* d_out, int out_size, void* d_ws, size_t ws_size,
                              hipStream_t stream) {
    (void)in_sizes; (void)n_in; (void)out_size; (void)ws_size;

    const float* x     = (const float*)d_in[0];
    const float* gamma = (const float*)d_in[1];
    const float* beta  = (const float*)d_in[2];
    const float* Wq    = (const float*)d_in[3];
    const float* bq    = (const float*)d_in[4];
    const float* Wk    = (const float*)d_in[5];
    const float* bk    = (const float*)d_in[6];
    const float* Wv    = (const float*)d_in[7];
    const float* bv    = (const float*)d_in[8];
    const float* Wp    = (const float*)d_in[9];
    const float* bp    = (const float*)d_in[10];

    const size_t B = 16, N = 1024, C = 512;

    // workspace carve-up (256B aligned)
    char* ws = (char*)d_ws;
    size_t off = 0;
    auto take = [&](size_t bytes) {
        size_t o = off;
        off += (bytes + 255) & ~(size_t)255;
        return o;
    };
    unsigned short* WqT   = (unsigned short*)(ws + take(C * C * 2));
    unsigned short* WkT   = (unsigned short*)(ws + take(C * C * 2));
    unsigned short* WvT   = (unsigned short*)(ws + take(C * C * 2));
    unsigned short* WpT   = (unsigned short*)(ws + take(C * C * 2));
    unsigned short* xnbf  = (unsigned short*)(ws + take(B * N * C * 2));
    unsigned short* qbf   = (unsigned short*)(ws + take(B * N * C * 2));
    unsigned short* kbf   = (unsigned short*)(ws + take(B * N * C * 2));
    unsigned short* vTbf  = (unsigned short*)(ws + take(B * C * N * 2));   // [b][c][m]
    unsigned short* outbf = (unsigned short*)(ws + take(B * N * C * 2));
    unsigned short* attn  = (unsigned short*)(ws + take(B * N * N * 2));
    float*          scr   = (float*)        (ws + take(B * N * N * 4));

    const dim3 blk(256);

    // 1) weights -> bf16, transposed [out][in]
    transpose_w_kernel<<<1024, blk, 0, stream>>>(Wq, WqT);
    transpose_w_kernel<<<1024, blk, 0, stream>>>(Wk, WkT);
    transpose_w_kernel<<<1024, blk, 0, stream>>>(Wv, WvT);
    transpose_w_kernel<<<1024, blk, 0, stream>>>(Wp, WpT);

    // 2) GroupNorm: fp32 xn -> d_out (residual base), bf16 xn -> ws
    groupnorm_kernel<<<512, blk, 0, stream>>>(x, gamma, beta, (float*)d_out, xnbf);

    // 3) Q = xn*WqT + bq ; K = xn*WkT + bk  (per-batch M=1024, N=512, K=512)
    const dim3 gQKV(N / 128, 512 / 128, B);
    gemm_bf16_wmma<0, true><<<gQKV, blk, 0, stream>>>(
        xnbf, N * C, (int)C, WqT, 0, (int)C, qbf, N * C, (int)C, bq, 1.0f, (int)C);
    gemm_bf16_wmma<0, true><<<gQKV, blk, 0, stream>>>(
        xnbf, N * C, (int)C, WkT, 0, (int)C, kbf, N * C, (int)C, bk, 1.0f, (int)C);
    // V stored transposed: vT[b][c][m]
    gemm_bf16_wmma<2, true><<<gQKV, blk, 0, stream>>>(
        xnbf, N * C, (int)C, WvT, 0, (int)C, vTbf, C * N, (int)N, bv, 1.0f, (int)C);

    // 4) scores = (q . k^T) * 1/sqrt(512) -> fp32   (M=N=1024, K=512)
    const dim3 gS(N / 128, N / 128, B);
    gemm_bf16_wmma<1, false><<<gS, blk, 0, stream>>>(
        qbf, N * C, (int)C, kbf, N * C, (int)C, scr, N * N, (int)N,
        nullptr, 0.044194173824159216f, (int)C);

    // 5) softmax rows -> bf16 attn
    softmax_kernel<<<B * N, blk, 0, stream>>>(scr, attn);

    // 6) out = attn @ v   (A=attn[N][N], Bt=vT[c][m] -> [N=512][K=1024])
    const dim3 gAV(N / 128, 512 / 128, B);
    gemm_bf16_wmma<0, false><<<gAV, blk, 0, stream>>>(
        attn, N * N, (int)N, vTbf, C * N, (int)N, outbf, N * C, (int)C,
        nullptr, 1.0f, (int)N);

    // 7) d_out += out @ WpT + bp   (residual accumulate into xn already in d_out)
    gemm_bf16_wmma<3, true><<<gAV, blk, 0, stream>>>(
        outbf, N * C, (int)C, WpT, 0, (int)C, d_out, N * C, (int)C, bp, 1.0f, (int)C);
}
